// MultiHeadAttention_41506563949074
// MI455X (gfx1250) — matrix operand order repaired and checked by reading, not compiled, not run on hardware
//
#include <hip/hip_runtime.h>
#include <hip/hip_bf16.h>

// ---------------------------------------------------------------------------
// MHA forward for B=2, S=2048, E=1024, H=16, D=64 on gfx1250 (MI455X).
// bf16 WMMA (v_wmma_f32_16x16x32_bf16) everywhere, fp32 accumulate/softmax.
// ---------------------------------------------------------------------------

#define EMB   1024
#define SEQ   2048
#define NB    2
#define NH    16
#define HD    64
#define NROWS (NB*SEQ)          // 4096

typedef __attribute__((ext_vector_type(8)))  float  v8f;
typedef __attribute__((ext_vector_type(16))) __bf16 v16bf;
typedef __attribute__((ext_vector_type(8)))  __bf16 v8bf;

union V16 { v16bf v; v8bf h[2]; };

__device__ __forceinline__ unsigned short f32_to_bf16(float f) {
    unsigned int u = __float_as_uint(f);
    u += 0x7FFFu + ((u >> 16) & 1u);          // round to nearest even
    return (unsigned short)(u >> 16);
}

__device__ __forceinline__ v8f vzero8() {
    v8f z = {0.f,0.f,0.f,0.f,0.f,0.f,0.f,0.f};
    return z;
}

// ---------------------------------------------------------------------------
// fp32 -> bf16 conversion, 4 elements / thread
// ---------------------------------------------------------------------------
__global__ void cvt_f32_bf16(const float* __restrict__ in,
                             unsigned short* __restrict__ out, int n4) {
    int i = blockIdx.x * blockDim.x + threadIdx.x;
    if (i < n4) {
        float4 v = ((const float4*)in)[i];
        unsigned int a = (unsigned int)f32_to_bf16(v.x) | ((unsigned int)f32_to_bf16(v.y) << 16);
        unsigned int b = (unsigned int)f32_to_bf16(v.z) | ((unsigned int)f32_to_bf16(v.w) << 16);
        ((uint2*)out)[i] = make_uint2(a, b);
    }
}

// ---------------------------------------------------------------------------
// NT GEMM: out[n][f] = sum_e A[n][e]*W[f][e] + bias[f]   (A:[4096,1024] bf16,
// W:[1024,1024] bf16).  Block = 4 waves, block tile 128x64, wave tile 32x64.
// mode 0: bf16 out, layout [B,H,S,D], scaled      (Q with 1/8 folded, K)
// mode 1: bf16 out, layout [B,H,D,S] (transposed) (V)
// mode 2: f32 out, row-major [N,F]                (final projection)
// ---------------------------------------------------------------------------
__global__ __launch_bounds__(128)
void gemm_nt(const unsigned short* __restrict__ A,
             const unsigned short* __restrict__ W,
             const float* __restrict__ bias,
             void* __restrict__ outp, float scale, int mode) {
    const int lane = threadIdx.x & 31;
    const int wave = threadIdx.x >> 5;
    const int half = lane >> 4;
    const int ln   = lane & 15;

    const int row0 = blockIdx.x * 128 + wave * 32;   // 32 row-blocks
    const int f0   = blockIdx.y * 64;                // 16 col-blocks

    v8f acc[2][4];
#pragma unroll
    for (int i = 0; i < 2; i++)
#pragma unroll
        for (int j = 0; j < 4; j++) acc[i][j] = vzero8();

    // A-fragment rows: lanes 0-15 and 16-31 hold the SAME rows, different K.
    const unsigned short* Arow0 = A + (size_t)(row0 + ln) * EMB;
    const unsigned short* Arow1 = A + (size_t)(row0 + 16 + ln) * EMB;
    const unsigned short* Wcol[4];
#pragma unroll
    for (int j = 0; j < 4; j++)
        Wcol[j] = W + (size_t)(f0 + 16 * j + ln) * EMB + 16 * half;

    for (int e0 = 0; e0 < EMB; e0 += 32) {
        if (e0 + 64 < EMB) {   // CDNA5 global_prefetch_b8 hints (L2-resident data)
            __builtin_prefetch(Arow0 + e0 + 64, 0, 3);
            __builtin_prefetch(Wcol[0] + e0 + 64, 0, 3);
        }
        V16 a0, a1;
        a0.h[0] = *(const v8bf*)(Arow0 + e0 + 16 * half);      // K 0..7 / 16..23
        a0.h[1] = *(const v8bf*)(Arow0 + e0 + 8 + 16 * half);  // K 8..15 / 24..31
        a1.h[0] = *(const v8bf*)(Arow1 + e0 + 16 * half);
        a1.h[1] = *(const v8bf*)(Arow1 + e0 + 8 + 16 * half);
        v16bf b[4];
#pragma unroll
        for (int j = 0; j < 4; j++)
            b[j] = *(const v16bf*)(Wcol[j] + e0);              // 32B contiguous
#pragma unroll
        for (int j = 0; j < 4; j++) {
            acc[0][j] = __builtin_amdgcn_wmma_f32_16x16x32_bf16(
                false, a0.v, false, b[j], (short)0, acc[0][j], false, false);
            acc[1][j] = __builtin_amdgcn_wmma_f32_16x16x32_bf16(
                false, a1.v, false, b[j], (short)0, acc[1][j], false, false);
        }
    }

    // Epilogue.  C layout: VGPR r, lane l -> row = r + 8*(l/16), col = l%16.
#pragma unroll
    for (int i = 0; i < 2; i++) {
#pragma unroll
        for (int j = 0; j < 4; j++) {
            const int f = f0 + 16 * j + ln;
            const float bcol = bias[f];
            if (mode == 1) {
                // V^T [B,H,D,S]: per-lane 8 consecutive s -> one 16B store
                const int n0 = row0 + 16 * i + 8 * half;
                const int bb = n0 >> 11, s0 = n0 & (SEQ - 1);
                const int hh = f >> 6, dd = f & 63;
                size_t base = (((size_t)(bb * NH + hh)) * HD + dd) * SEQ + s0;
                unsigned int u[4];
#pragma unroll
                for (int p = 0; p < 4; p++) {
                    unsigned short lo = f32_to_bf16((acc[i][j][2 * p] + bcol) * scale);
                    unsigned short hi = f32_to_bf16((acc[i][j][2 * p + 1] + bcol) * scale);
                    u[p] = (unsigned int)lo | ((unsigned int)hi << 16);
                }
                uint4 pk; pk.x = u[0]; pk.y = u[1]; pk.z = u[2]; pk.w = u[3];
                *(uint4*)((unsigned short*)outp + base) = pk;
            } else {
#pragma unroll
                for (int r = 0; r < 8; r++) {
                    const int n = row0 + 16 * i + r + 8 * half;
                    const float val = (acc[i][j][r] + bcol) * scale;
                    if (mode == 0) {
                        const int bb = n >> 11, s = n & (SEQ - 1);
                        const int hh = f >> 6, dd = f & 63;
                        size_t idx = (((size_t)(bb * NH + hh)) * SEQ + s) * (size_t)HD + dd;
                        ((unsigned short*)outp)[idx] = f32_to_bf16(val);
                    } else {
                        ((float*)outp)[(size_t)n * EMB + f] = val;
                    }
                }
            }
        }
    }
}

// ---------------------------------------------------------------------------
// Flash-style causal attention.  Q,K in [B,H,S,D] bf16 (Q pre-scaled 1/8),
// V in [B,H,D,S] bf16.  One wave per (b,h, 16-query tile); keys in blocks of
// 32 with online softmax.  ctx out bf16 [B,S,E].
// ---------------------------------------------------------------------------
__global__ __launch_bounds__(128)
void attention_fwd(const unsigned short* __restrict__ Q,
                   const unsigned short* __restrict__ K,
                   const unsigned short* __restrict__ V,
                   unsigned short* __restrict__ CTX) {
    __shared__ unsigned short pbuf[4][16 * 32];   // per-wave P staging (1KB)

    const int lane = threadIdx.x & 31;
    const int wave = threadIdx.x >> 5;
    const int half = lane >> 4;
    const int ln   = lane & 15;

    const int unit = blockIdx.x * 4 + wave;       // 4096 units
    const int bh = unit >> 7;                     // b*16+h
    const int q0 = (unit & 127) * 16;

    const unsigned short* Qb = Q + (size_t)bh * SEQ * HD;
    const unsigned short* Kb = K + (size_t)bh * SEQ * HD;
    const unsigned short* Vt = V + (size_t)bh * HD * SEQ;

    // Q A-fragments (contraction = d): frag0 d0..31, frag1 d32..63
    const unsigned short* qrow = Qb + (size_t)(q0 + ln) * HD;
    V16 qa[2];
    qa[0].h[0] = *(const v8bf*)(qrow + 16 * half);
    qa[0].h[1] = *(const v8bf*)(qrow + 8 + 16 * half);
    qa[1].h[0] = *(const v8bf*)(qrow + 32 + 16 * half);
    qa[1].h[1] = *(const v8bf*)(qrow + 40 + 16 * half);

    v8f acc[4];
#pragma unroll
    for (int t = 0; t < 4; t++) acc[t] = vzero8();
    float mrow[8], lrow[8];
#pragma unroll
    for (int r = 0; r < 8; r++) { mrow[r] = -3.0e38f; lrow[r] = 0.f; }

    unsigned short* pb = &pbuf[wave][0];
    const int lastkey = q0 + 15;

    for (int j = 0; j <= lastkey; j += 32) {
        // ---- scores for two 16-key subtiles (contraction d = 64 -> 2 WMMAs each)
        v8f sc[2];
#pragma unroll
        for (int t = 0; t < 2; t++) {
            const unsigned short* krow = Kb + (size_t)(j + 16 * t + ln) * HD;
            v16bf kb0 = *(const v16bf*)(krow + 16 * half);        // d 0..31
            v16bf kb1 = *(const v16bf*)(krow + 32 + 16 * half);   // d 32..63
            v8f c = vzero8();
            c = __builtin_amdgcn_wmma_f32_16x16x32_bf16(false, qa[0].v, false, kb0,
                                                        (short)0, c, false, false);
            c = __builtin_amdgcn_wmma_f32_16x16x32_bf16(false, qa[1].v, false, kb1,
                                                        (short)0, c, false, false);
            sc[t] = c;
        }

        // ---- causal mask
#pragma unroll
        for (int t = 0; t < 2; t++) {
            const int key = j + 16 * t + ln;
#pragma unroll
            for (int r = 0; r < 8; r++) {
                const int qi = q0 + r + 8 * half;
                if (key > qi) sc[t][r] = -3.0e38f;
            }
        }

        // ---- online softmax (row = r + 8*half spans lanes sharing (lane&16))
        float rmax[8];
#pragma unroll
        for (int r = 0; r < 8; r++) rmax[r] = fmaxf(sc[0][r], sc[1][r]);
#pragma unroll
        for (int m = 1; m < 16; m <<= 1)
#pragma unroll
            for (int r = 0; r < 8; r++)
                rmax[r] = fmaxf(rmax[r], __shfl_xor(rmax[r], m, 32));

        float alpha[8], rsum[8];
#pragma unroll
        for (int r = 0; r < 8; r++) {
            float mn = fmaxf(mrow[r], rmax[r]);
            alpha[r] = __expf(mrow[r] - mn);
            mrow[r] = mn;
            float p0 = __expf(sc[0][r] - mn);
            float p1 = __expf(sc[1][r] - mn);
            sc[0][r] = p0; sc[1][r] = p1;
            rsum[r] = p0 + p1;
        }
#pragma unroll
        for (int m = 1; m < 16; m <<= 1)
#pragma unroll
            for (int r = 0; r < 8; r++)
                rsum[r] += __shfl_xor(rsum[r], m, 32);
#pragma unroll
        for (int r = 0; r < 8; r++) lrow[r] = lrow[r] * alpha[r] + rsum[r];
#pragma unroll
        for (int t = 0; t < 4; t++)
#pragma unroll
            for (int r = 0; r < 8; r++) acc[t][r] *= alpha[r];

        // ---- relayout P: C-layout -> A-fragment via per-wave LDS
#pragma unroll
        for (int t = 0; t < 2; t++)
#pragma unroll
            for (int r = 0; r < 8; r++)
                pb[(r + 8 * half) * 32 + 16 * t + ln] = f32_to_bf16(sc[t][r]);
        asm volatile("s_wait_dscnt 0x0" ::: "memory");
        V16 pa;
        pa.h[0] = *(const v8bf*)(pb + ln * 32 + 16 * half);
        pa.h[1] = *(const v8bf*)(pb + ln * 32 + 8 + 16 * half);
        asm volatile("" ::: "memory");

        // ---- PV: contraction = 32 keys, 4 d-tiles
#pragma unroll
        for (int t = 0; t < 4; t++) {
            v16bf vb = *(const v16bf*)(Vt + (size_t)(16 * t + ln) * SEQ + j + 16 * half);
            acc[t] = __builtin_amdgcn_wmma_f32_16x16x32_bf16(false, pa.v, false, vb,
                                                             (short)0, acc[t], false, false);
        }
    }

    // ---- normalize + store ctx bf16 [B,S,E]
    float inv[8];
#pragma unroll
    for (int r = 0; r < 8; r++) inv[r] = 1.0f / lrow[r];
    const int bb = bh >> 4, hh = bh & 15;
#pragma unroll
    for (int t = 0; t < 4; t++)
#pragma unroll
        for (int r = 0; r < 8; r++) {
            const int qi = q0 + r + 8 * half;
            const int dd = 16 * t + ln;
            size_t idx = ((size_t)bb * SEQ + qi) * EMB + hh * 64 + dd;
            CTX[idx] = f32_to_bf16(acc[t][r] * inv[r]);
        }
}

// ---------------------------------------------------------------------------
extern "C" void kernel_launch(void* const* d_in, const int* in_sizes, int n_in,
                              void* d_out, int out_size, void* d_ws, size_t ws_size,
                              hipStream_t stream) {
    (void)in_sizes; (void)n_in; (void)out_size; (void)ws_size;

    const float* x  = (const float*)d_in[0];
    const float* Wq = (const float*)d_in[1];
    const float* bq = (const float*)d_in[2];
    const float* Wk = (const float*)d_in[3];
    const float* bk = (const float*)d_in[4];
    const float* Wv = (const float*)d_in[5];
    const float* bv = (const float*)d_in[6];
    const float* Wo = (const float*)d_in[7];
    const float* bo = (const float*)d_in[8];
    float* out = (float*)d_out;

    // workspace layout (48 MB total, bf16 elements)
    unsigned short* xb  = (unsigned short*)d_ws;                 // 4M
    unsigned short* wqb = xb  + (size_t)NROWS * EMB;             // 1M
    unsigned short* wkb = wqb + (size_t)EMB * EMB;
    unsigned short* wvb = wkb + (size_t)EMB * EMB;
    unsigned short* wob = wvb + (size_t)EMB * EMB;
    unsigned short* Qw  = wob + (size_t)EMB * EMB;               // 4M each
    unsigned short* Kw  = Qw  + (size_t)NROWS * EMB;
    unsigned short* Vw  = Kw  + (size_t)NROWS * EMB;
    unsigned short* CT  = Vw  + (size_t)NROWS * EMB;

    // 1) fp32 -> bf16 conversions
    {
        int n4 = (NROWS * EMB) / 4;
        cvt_f32_bf16<<<(n4 + 255) / 256, 256, 0, stream>>>(x, xb, n4);
        int w4 = (EMB * EMB) / 4;
        cvt_f32_bf16<<<(w4 + 255) / 256, 256, 0, stream>>>(Wq, wqb, w4);
        cvt_f32_bf16<<<(w4 + 255) / 256, 256, 0, stream>>>(Wk, wkb, w4);
        cvt_f32_bf16<<<(w4 + 255) / 256, 256, 0, stream>>>(Wv, wvb, w4);
        cvt_f32_bf16<<<(w4 + 255) / 256, 256, 0, stream>>>(Wo, wob, w4);
    }

    dim3 ggrid(NROWS / 128, EMB / 64);
    // 2) projections (1/sqrt(64)=0.125 folded into Q)
    gemm_nt<<<ggrid, 128, 0, stream>>>(xb, wqb, bq, Qw, 0.125f, 0);
    gemm_nt<<<ggrid, 128, 0, stream>>>(xb, wkb, bk, Kw, 1.0f,   0);
    gemm_nt<<<ggrid, 128, 0, stream>>>(xb, wvb, bv, Vw, 1.0f,   1);

    // 3) causal attention: 32 (b,h) * 128 q-tiles = 4096 wave-units
    attention_fwd<<<(NB * NH * (SEQ / 16)) / 4, 128, 0, stream>>>(Qw, Kw, Vw, CT);

    // 4) output projection -> fp32 d_out
    gemm_nt<<<ggrid, 128, 0, stream>>>(CT, wob, bo, out, 1.0f, 2);
}
